// IPLM_84318797955723
// MI455X (gfx1250) — matrix operand order repaired and checked by reading, not compiled, not run on hardware
//
#include <hip/hip_runtime.h>
#include <hip/hip_bf16.h>

#define D        512
#define KSZ      4096
#define NTOK     32768           // 8 * 4096
#define MBLK     128
#define NBLK     64
#define NPANELS  (KSZ / NBLK)    // 64
#define NTILES   4               // 16-wide n-tiles per wave
#define KSTEP    32
#define EMA      0.99f
#define EPSV     1e-5f

typedef __attribute__((ext_vector_type(16))) __bf16 v16bf;
typedef __attribute__((ext_vector_type(8)))  float  v8f;
typedef __attribute__((ext_vector_type(4)))  unsigned int v4u;
typedef __attribute__((ext_vector_type(8)))  int    v8i_t;
typedef __attribute__((ext_vector_type(4)))  int    v4i_t;

#if __has_builtin(__builtin_amdgcn_tensor_load_to_lds) && \
    __has_builtin(__builtin_amdgcn_s_wait_tensorcnt)
#define USE_TDM 1
#else
#define USE_TDM 0
#endif

// LDS layout (bytes) inside dynamic shared memory
#define OFF_SK0  (MBLK * D * 2)               // 131072
#define OFF_SK1  (OFF_SK0 + NBLK * D * 2)     // 196608
#define OFF_RED  (OFF_SK1 + NBLK * D * 2)     // 262144
#define SMEM_SZ  (OFF_RED + MBLK * 16 * 8)    // 278528 B = 272 KB < 320 KB/WGP

__device__ __forceinline__ unsigned int f2bf(float f) {
    union { float f; unsigned int u; } x; x.f = f;
    unsigned int u = x.u;
    return (u + 0x7FFFu + ((u >> 16) & 1u)) >> 16;   // round-to-nearest-even
}

union ABFrag { uint4 q[2]; v16bf v; };

#if USE_TDM
// Issue one TDM copy: 64x512 bf16 codebook panel (64 KB) -> LDS.
// D# per CDNA5 ISA: group0 = {count=1 | lds_addr |
// global_addr | type=2}; group1 = {data_size=8B, tensor_dim0, tile_dim0(1-D),
// tensor_dim0_stride}; groups 2/3 zero (<=2-D tensor).
__device__ __forceinline__ void tdm_load_panel(unsigned long long ga,
                                               unsigned lds_off) {
    v4u g0;
    g0[0] = 1u;                                        // count=1, no gather
    g0[1] = lds_off;                                   // lds_addr (bytes)
    g0[2] = (unsigned)(ga & 0xFFFFFFFFull);            // global_addr[31:0]
    g0[3] = (unsigned)((ga >> 32) & 0x01FFFFFFull)     // global_addr[56:32]
            | 0x80000000u;                             // type = 2 ("image")
    const unsigned td0 = (unsigned)((KSZ * D * 2) / 8);   // tensor len, 8B units
    const unsigned tl0 = (unsigned)((NBLK * D * 2) / 8);  // tile len = 8192
    v8i_t g1;
    g1[0] = (int)(3u << 16);                  // workgroup_mask=0, data_size=8B
    g1[1] = (int)((td0 & 0xFFFFu) << 16);     // tensor_dim0[15:0]
    g1[2] = (int)((td0 >> 16) & 0xFFFFu);     // tensor_dim0[31:16], dim1=0
    g1[3] = (int)((tl0 & 0xFFFFu) << 16);     // tile_dim0 (1-D tile)
    g1[4] = 0;                                // tile_dim1/2 unused
    g1[5] = (int)td0;                         // tensor_dim0_stride[31:0]
    g1[6] = 0;
    g1[7] = 0;
    v4i_t gz = {0, 0, 0, 0};
#if defined(__clang_major__) && (__clang_major__ >= 23)
    v8i_t gz8 = {0, 0, 0, 0, 0, 0, 0, 0};
    __builtin_amdgcn_tensor_load_to_lds(g0, g1, gz, gz, gz8, 0);
#else
    __builtin_amdgcn_tensor_load_to_lds(g0, g1, gz, gz, 0);
#endif
}
#endif

// ---------------------------------------------------------------------------
// K0: initialize output accumulators (EMA-scaled) and scalars
// ---------------------------------------------------------------------------
__global__ void k0_init(const float* __restrict__ ecs_in,
                        const float* __restrict__ emaw_in,
                        float* __restrict__ o_ecs,
                        float* __restrict__ o_emaw,
                        float* __restrict__ o_loss,
                        float* __restrict__ nsum) {
    int i = blockIdx.x * 256 + threadIdx.x;
    if (i < KSZ * D) o_emaw[i] = EMA * emaw_in[i];
    if (i < KSZ)     o_ecs[i]  = EMA * ecs_in[i];
    if (i == 0) { o_loss[0] = 0.0f; nsum[0] = 0.0f; }
}

// ---------------------------------------------------------------------------
// K1: codebook prep — bf16 copy of K (packed 2 per dword) + fp32 row norms
// ---------------------------------------------------------------------------
__global__ __launch_bounds__(256) void k1_prep(const float* __restrict__ Kin,
                                               unsigned int* __restrict__ Kb,
                                               float* __restrict__ Knorm) {
    __shared__ float s[256];
    int k = blockIdx.x, t = threadIdx.x;
    float2 v = ((const float2*)(Kin + k * D))[t];
    Kb[k * (D / 2) + t] = f2bf(v.x) | (f2bf(v.y) << 16);
    s[t] = v.x * v.x + v.y * v.y;
    __syncthreads();
    for (int o = 128; o > 0; o >>= 1) {
        if (t < o) s[t] += s[t + o];
        __syncthreads();
    }
    if (t == 0) Knorm[k] = s[0];
}

// ---------------------------------------------------------------------------
// K2: WMMA GEMM (f · K^T) + fused per-row argmin of (||k||^2 - 2 f·k)
//   256 threads = 8 waves; wave w owns M rows [w*16, w*16+16)
//   f panel 128x512 bf16 LDS-resident; K panels double-buffered via TDM
// ---------------------------------------------------------------------------
__global__ __launch_bounds__(256) void k2_gemm_argmin(
        const float* __restrict__ f,
        const uint4* __restrict__ Kb,          // bf16 codebook, 8 elems / uint4
        const float* __restrict__ Knorm,
        int* __restrict__ k1_idx) {
    extern __shared__ char smem[];
    unsigned short* sF  = (unsigned short*)smem;             // 128*512 bf16
    unsigned short* sK0 = (unsigned short*)(smem + OFF_SK0); //  64*512 bf16
    unsigned short* sK1 = (unsigned short*)(smem + OFF_SK1); //  64*512 bf16
    float* sVal = (float*)(smem + OFF_RED);                  // 128*16
    int*   sIdx = (int*)(sVal + MBLK * 16);                  // 128*16

    const int tid  = threadIdx.x;
    const int w    = tid >> 5;
    const int lane = tid & 31;
    const int row0 = blockIdx.x * MBLK;

#if USE_TDM
    const unsigned long long kbBase = (unsigned long long)(size_t)Kb;
    const unsigned ldsK0 = (unsigned)(size_t)(void*)sK0;     // runtime value
    if (w == 0) tdm_load_panel(kbBase, ldsK0);               // prefetch panel 0
#endif

    // ---- stage f block -> bf16 in LDS (each element read from HBM once) ----
    {
        const float4* f4 = (const float4*)(f + (size_t)row0 * D);
        uint2* s2 = (uint2*)sF;
        for (int i = tid; i < MBLK * D / 4; i += 256) {
            float4 v = f4[i];
            uint2 p;
            p.x = f2bf(v.x) | (f2bf(v.y) << 16);
            p.y = f2bf(v.z) | (f2bf(v.w) << 16);
            s2[i] = p;
        }
    }

    // per-lane fragment coordinates (ISA 16-bit A / B wave32 layouts)
    const int mrow  = w * 16 + (lane & 15);   // A: lane -> M row
    const int khalf = lane >> 4;              // A: which K-half of the row
    const int ncol  = lane & 15;              // B/C/D: lane -> N column
    const int ksel  = lane >> 4;              // B: which K-half

    float bestv[8];
    int   besti[8];
#pragma unroll
    for (int r = 0; r < 8; ++r) { bestv[r] = 3.4e38f; besti[r] = 0; }

    for (int nb = 0; nb < NPANELS; ++nb) {
#if USE_TDM
        if (w == 0) __builtin_amdgcn_s_wait_tensorcnt(0);
        __syncthreads();                       // panel nb visible to all waves
        if (w == 0 && nb + 1 < NPANELS)        // overlap next copy with compute
            tdm_load_panel(
                kbBase + (unsigned long long)(nb + 1) * (NBLK * D * 2),
                ldsK0 + (unsigned)(((nb + 1) & 1) * (NBLK * D * 2)));
        const unsigned short* sK = (nb & 1) ? sK1 : sK0;
#else
        __syncthreads();
        {
            const uint4* kp = Kb + (size_t)nb * (NBLK * D / 8);
            uint4* sk4 = (uint4*)sK0;
            for (int i = tid; i < NBLK * D / 8; i += 256) sk4[i] = kp[i];
        }
        __syncthreads();
        const unsigned short* sK = sK0;
#endif

        v8f acc[NTILES];
#pragma unroll
        for (int nt = 0; nt < NTILES; ++nt)
            acc[nt] = (v8f){0.f, 0.f, 0.f, 0.f, 0.f, 0.f, 0.f, 0.f};

#pragma unroll 4
        for (int ks = 0; ks < D / KSTEP; ++ks) {
            const int k0 = ks * KSTEP;
            ABFrag a;
            const uint4* ap = (const uint4*)(sF + mrow * D + k0 + khalf * 8);
            a.q[0] = ap[0];          // K = k0 + khalf*8      .. +8
            a.q[1] = ap[2];          // K = k0 + 16 + khalf*8 .. +8
            ABFrag b[NTILES];        // load ALL B frags, then WMMA back-to-back
#pragma unroll
            for (int nt = 0; nt < NTILES; ++nt) {
                const uint4* bp =
                    (const uint4*)(sK + (nt * 16 + ncol) * D + k0 + ksel * 8);
                b[nt].q[0] = bp[0];
                b[nt].q[1] = bp[2];
            }
#pragma unroll
            for (int nt = 0; nt < NTILES; ++nt)
                acc[nt] = __builtin_amdgcn_wmma_f32_16x16x32_bf16(
                    false, a.v, false, b[nt].v, (short)0, acc[nt], false, false);
        }

        // fold distances into per-lane running argmin
#pragma unroll
        for (int nt = 0; nt < NTILES; ++nt) {
            const int n = nb * NBLK + nt * 16 + ncol;
            const float kn = Knorm[n];
#pragma unroll
            for (int r = 0; r < 8; ++r) {
                float dval = kn - 2.0f * acc[nt][r];
                if (dval < bestv[r]) { bestv[r] = dval; besti[r] = n; }
            }
        }
    }

    // ---- cross-lane argmin: 16 candidate lanes per row ----
    __syncthreads();
#pragma unroll
    for (int r = 0; r < 8; ++r) {
        int mloc = w * 16 + r + 8 * (lane >> 4);   // C/D layout row
        sVal[mloc * 16 + ncol] = bestv[r];
        sIdx[mloc * 16 + ncol] = besti[r];
    }
    __syncthreads();
    if (tid < MBLK) {
        float bv = sVal[tid * 16];
        int   bi = sIdx[tid * 16];
        for (int c = 1; c < 16; ++c) {
            float v = sVal[tid * 16 + c];
            int   ii = sIdx[tid * 16 + c];
            if (v < bv || (v == bv && ii < bi)) { bv = v; bi = ii; }
        }
        k1_idx[row0 + tid] = bi;
    }
}

// ---------------------------------------------------------------------------
// K3: gather K[k1] -> f_ipmlm, loss partials, EMA scatter (atomics)
// ---------------------------------------------------------------------------
__global__ __launch_bounds__(256) void k3_gather_scatter(
        const float* __restrict__ f,
        const float* __restrict__ Kin,
        const int* __restrict__ k1_idx,
        float* __restrict__ o_f,
        float* __restrict__ o_loss,
        float* __restrict__ o_ecs,
        float* __restrict__ o_emaw) {
    __shared__ float s[256];
    const int n = blockIdx.x, t = threadIdx.x;
    const int k = k1_idx[n];
    float2 fv = ((const float2*)(f + (size_t)n * D))[t];
    float2 kv = ((const float2*)(Kin + (size_t)k * D))[t];
    ((float2*)(o_f + (size_t)n * D))[t] = kv;
    float dx = kv.x - fv.x, dy = kv.y - fv.y;
    s[t] = dx * dx + dy * dy;
    atomicAdd(&o_emaw[(size_t)k * D + 2 * t],     (1.0f - EMA) * fv.x);
    atomicAdd(&o_emaw[(size_t)k * D + 2 * t + 1], (1.0f - EMA) * fv.y);
    __syncthreads();
    for (int o = 128; o > 0; o >>= 1) {
        if (t < o) s[t] += s[t + o];
        __syncthreads();
    }
    if (t == 0) {
        atomicAdd(o_loss, s[0]);
        atomicAdd(&o_ecs[k], 1.0f - EMA);
    }
}

// ---------------------------------------------------------------------------
// K4: n = sum(new_ecs); finalize loss (mean)
// ---------------------------------------------------------------------------
__global__ void k4_reduce(const float* __restrict__ o_ecs,
                          float* __restrict__ nsum,
                          float* __restrict__ o_loss) {
    __shared__ float s[256];
    int t = threadIdx.x;
    float a = 0.0f;
    for (int i = t; i < KSZ; i += 256) a += o_ecs[i];
    s[t] = a;
    __syncthreads();
    for (int o = 128; o > 0; o >>= 1) {
        if (t < o) s[t] += s[t + o];
        __syncthreads();
    }
    if (t == 0) {
        nsum[0] = s[0];
        o_loss[0] = o_loss[0] * (1.0f / ((float)NTOK * (float)D));
    }
}

// ---------------------------------------------------------------------------
// K5: new_K = new_ema_w / smoothed
// ---------------------------------------------------------------------------
__global__ __launch_bounds__(256) void k5_newK(const float* __restrict__ o_ecs,
                                               const float* __restrict__ o_emaw,
                                               const float* __restrict__ nsum,
                                               float* __restrict__ o_K) {
    const int k = blockIdx.x, t = threadIdx.x;
    const float n = nsum[0];
    const float sm = (o_ecs[k] + EPSV) / (n + (float)KSZ * EPSV) * n;
    const float inv = 1.0f / sm;
    float2 wv = ((const float2*)(o_emaw + (size_t)k * D))[t];
    float2 r; r.x = wv.x * inv; r.y = wv.y * inv;
    ((float2*)(o_K + (size_t)k * D))[t] = r;
}

// ---------------------------------------------------------------------------
extern "C" void kernel_launch(void* const* d_in, const int* in_sizes, int n_in,
                              void* d_out, int out_size, void* d_ws, size_t ws_size,
                              hipStream_t stream) {
    const float* f       = (const float*)d_in[0];   // [8,4096,512]
    const float* Kin     = (const float*)d_in[1];   // [4096,512]
    const float* ecs_in  = (const float*)d_in[2];   // [4096]
    const float* emaw_in = (const float*)d_in[3];   // [4096,512]

    float* out      = (float*)d_out;
    float* o_f      = out;                               // 16777216
    float* o_loss   = out + (size_t)NTOK * D;            // 1
    float* o_K      = o_loss + 1;                        // 2097152
    float* o_ecs    = o_K + (size_t)KSZ * D;             // 4096
    float* o_emaw   = o_ecs + KSZ;                       // 2097152

    char* ws = (char*)d_ws;
    unsigned int* Kb   = (unsigned int*)ws;              // 4 MB packed bf16
    float*        Kn   = (float*)(ws + (size_t)KSZ * D * 2);
    int*          kidx = (int*)((char*)Kn + KSZ * sizeof(float));
    float*        nsum = (float*)((char*)kidx + NTOK * sizeof(int));

    (void)hipFuncSetAttribute((const void*)k2_gemm_argmin,
                              hipFuncAttributeMaxDynamicSharedMemorySize,
                              SMEM_SZ);

    k0_init<<<(KSZ * D + 255) / 256, 256, 0, stream>>>(ecs_in, emaw_in,
                                                       o_ecs, o_emaw, o_loss, nsum);
    k1_prep<<<KSZ, 256, 0, stream>>>(Kin, Kb, Kn);
    k2_gemm_argmin<<<NTOK / MBLK, 256, SMEM_SZ, stream>>>(f, (const uint4*)Kb,
                                                          Kn, kidx);
    k3_gather_scatter<<<NTOK, 256, 0, stream>>>(f, Kin, kidx,
                                                o_f, o_loss, o_ecs, o_emaw);
    k4_reduce<<<1, 256, 0, stream>>>(o_ecs, nsum, o_loss);
    k5_newK<<<KSZ, 256, 0, stream>>>(o_ecs, o_emaw, nsum, o_K);
}
